// DecoderBlock_43044162241285
// MI455X (gfx1250) — matrix operand order repaired and checked by reading, not compiled, not run on hardware
//
#include <hip/hip_runtime.h>
#include <hip/hip_bf16.h>

// ---------------------------------------------------------------------------
// Types for CDNA5 WMMA (wave32)
// ---------------------------------------------------------------------------
typedef __attribute__((ext_vector_type(16))) __bf16 v16bf;
typedef __attribute__((ext_vector_type(8)))  float  v8f;

union FragBF {
  v16bf v;
  uint4 u[2];
};

__device__ __forceinline__ unsigned short f2bf(float x) {
  unsigned int u = __builtin_bit_cast(unsigned int, x);
  unsigned int r = (u >> 16) & 1u;
  u += 0x7fffu + r;                 // round-to-nearest-even
  return (unsigned short)(u >> 16);
}

// ---------------------------------------------------------------------------
// fp32 -> bf16 conversion (vectorized)
// ---------------------------------------------------------------------------
__global__ void cvt_f32_bf16(const float* __restrict__ in,
                             unsigned short* __restrict__ out, int n4) {
  int i = blockIdx.x * blockDim.x + threadIdx.x;
  if (i < n4) {
    float4 f = ((const float4*)in)[i];
    ushort4 o;
    o.x = f2bf(f.x); o.y = f2bf(f.y); o.z = f2bf(f.z); o.w = f2bf(f.w);
    ((ushort4*)out)[i] = o;
  }
}

// ---------------------------------------------------------------------------
// GEMM: C[M,N] = act(scale * (A[M,K](bf16) @ B[N,K](bf16)^T) + bias)
// Block = 256 threads = 8 waves arranged 4x2; tile 128x128, BK=32.
// Each wave: 32 rows x 64 cols (2 A-frags x 4 B-frags -> 8 WMMA / K-step).
// Double-buffered LDS staging: global loads of step k+1 overlap WMMA of k.
// ---------------------------------------------------------------------------
template <bool BIAS, bool RELU, bool OBF>
__global__ __launch_bounds__(256, 1)
void gemm_bt_kernel(const unsigned short* __restrict__ A, int lda,
                    const unsigned short* __restrict__ Bw, int ldb,
                    const float* __restrict__ bias,
                    void* __restrict__ Cv, int ldc, int K, float scale) {
  __shared__ unsigned short As[2][128 * 40];   // stride 40 = bank-spread pad
  __shared__ unsigned short Bs[2][128 * 40];

  const int tid  = threadIdx.x;
  const int w    = tid >> 5;
  const int lane = tid & 31;
  const int hf   = lane >> 4;       // 16-lane half
  const int l16  = lane & 15;
  const int wr   = w >> 1;          // 0..3 : 32-row group
  const int wc   = w & 1;           // 0..1 : 64-col group
  const size_t m0 = (size_t)blockIdx.y * 128;
  const size_t n0 = (size_t)blockIdx.x * 128;

  const int r_st  = tid >> 2;       // staging row (0..63), +64 for i=1
  const int c8_st = (tid & 3) * 8;  // staging col (bf16 units)

  uint4 ra[2], rb[2];
  // prologue: load + store K-step 0
#pragma unroll
  for (int i = 0; i < 2; i++) {
    int r = r_st + i * 64;
    ra[i] = *(const uint4*)(A  + (m0 + r) * lda + c8_st);
    rb[i] = *(const uint4*)(Bw + (n0 + r) * ldb + c8_st);
  }
#pragma unroll
  for (int i = 0; i < 2; i++) {
    int r = r_st + i * 64;
    *(uint4*)&As[0][r * 40 + c8_st] = ra[i];
    *(uint4*)&Bs[0][r * 40 + c8_st] = rb[i];
  }
  __syncthreads();

  v8f acc[2][4] = {};

  for (int kb = 0; kb < K; kb += 32) {
    const int cur  = (kb >> 5) & 1;
    const bool more = (kb + 32) < K;
    if (more) {                      // prefetch next K-step into VGPRs
#pragma unroll
      for (int i = 0; i < 2; i++) {
        int r = r_st + i * 64;
        ra[i] = *(const uint4*)(A  + (m0 + r) * lda + kb + 32 + c8_st);
        rb[i] = *(const uint4*)(Bw + (n0 + r) * ldb + kb + 32 + c8_st);
      }
    }

    // A fragments: lanes<16: K0-7 & 16-23; lanes>=16: K8-15 & 24-31
    FragBF af[2];
#pragma unroll
    for (int rs = 0; rs < 2; rs++) {
      const unsigned short* p = &As[cur][(wr * 32 + rs * 16 + l16) * 40];
      af[rs].u[0] = *(const uint4*)(p + hf * 8);
      af[rs].u[1] = *(const uint4*)(p + 16 + hf * 8);
    }
#pragma unroll
    for (int nt = 0; nt < 4; nt++) {
      FragBF bf;
      const unsigned short* p = &Bs[cur][(wc * 64 + nt * 16 + l16) * 40 + hf * 8];
      bf.u[0] = *(const uint4*)p;
      bf.u[1] = *(const uint4*)(p + 16);
#pragma unroll
      for (int rs = 0; rs < 2; rs++)
        acc[rs][nt] = __builtin_amdgcn_wmma_f32_16x16x32_bf16(
            false, af[rs].v, false, bf.v, (short)0, acc[rs][nt], false, false);
    }

    if (more) {
#pragma unroll
      for (int i = 0; i < 2; i++) {
        int r = r_st + i * 64;
        *(uint4*)&As[cur ^ 1][r * 40 + c8_st] = ra[i];
        *(uint4*)&Bs[cur ^ 1][r * 40 + c8_st] = rb[i];
      }
      __syncthreads();
    }
  }

  // epilogue: C layout row = j + 8*hf, col = l16 within 16x16 tile
#pragma unroll
  for (int rs = 0; rs < 2; rs++) {
#pragma unroll
    for (int nt = 0; nt < 4; nt++) {
#pragma unroll
      for (int j = 0; j < 8; j++) {
        size_t row = m0 + wr * 32 + rs * 16 + j + 8 * hf;
        size_t col = n0 + wc * 64 + nt * 16 + l16;
        float v = acc[rs][nt][j] * scale;
        if (BIAS) v += bias[col];
        if (RELU) v = fmaxf(v, 0.f);
        if (OBF) ((unsigned short*)Cv)[row * ldc + col] = f2bf(v);
        else     ((float*)Cv)[row * ldc + col] = v;
      }
    }
  }
}

// ---------------------------------------------------------------------------
// Fused (flash-style) attention, dk = 64.  Q is pre-scaled by 1/sqrt(dk).
// grid = (T/128, B*H), 256 threads (8 waves). Each wave owns 16 query rows.
// Scores stay in WMMA accumulators; P round-trips through per-wave LDS.
// Mask for the whole 128-query block is packed once into a 16 KB LDS bitmask;
// per tile each row needs a single 16-B ds_load for its 128 mask bits.
// K/V tiles are register-prefetched so global loads overlap the WMMA chain.
// ---------------------------------------------------------------------------
__global__ __launch_bounds__(256, 1)
void attn_fused_kernel(const unsigned short* __restrict__ Qg,
                       const unsigned short* __restrict__ Kg,
                       const unsigned short* __restrict__ Vg,
                       const int* __restrict__ mask,
                       unsigned short* __restrict__ Og,
                       int T, int D, int H) {
  extern __shared__ unsigned short smem[];
  unsigned short* Qs = smem;                 // 128 x 72  (Q tile, dim-contig)
  unsigned short* Ks = Qs + 128 * 72;        // 128 x 72  (K tile, dim-contig)
  unsigned short* Vs = Ks + 128 * 72;        // 64  x 136 (V^T: dim-major)
  unsigned short* Ps = Vs + 64 * 136;        // 128 x 136 (P, per-wave rows)
  unsigned int*   Ms = (unsigned int*)(Ps + 128 * 136);  // 128 x 32 dwords

  const int tid  = threadIdx.x;
  const int w    = tid >> 5;
  const int lane = tid & 31;
  const int hf   = lane >> 4;
  const int l16  = lane & 15;
  const int b  = blockIdx.y / H;
  const int h  = blockIdx.y % H;
  const int q0 = blockIdx.x * 128;
  const size_t qbase  = ((size_t)b * T + q0) * D + (size_t)h * 64;
  const size_t kvbase = ((size_t)b * T) * D + (size_t)h * 64;
  const int* mb = mask + (size_t)b * T * T;

  // stage Q tile (128 x 64 bf16)
#pragma unroll
  for (int i = 0; i < 4; i++) {
    int ii = tid + i * 256;              // 1024 uint4
    int r = ii >> 3, c8 = (ii & 7) * 8;
    *(uint4*)&Qs[r * 72 + c8] = *(const uint4*)(Qg + qbase + (size_t)r * D + c8);
  }
  // pack full-row mask bitmask once: thread -> (row, 512-col half)
  {
    int r = tid >> 1, hcol = tid & 1;
    const uint4* mp = (const uint4*)(mb + (size_t)(q0 + r) * T + hcol * 512);
#pragma unroll
    for (int i = 0; i < 16; i++) {
      unsigned int bits = 0;
#pragma unroll
      for (int k = 0; k < 8; k++) {
        uint4 m4 = mp[i * 8 + k];
        bits |= (m4.x != 0u ? 1u : 0u) << (k * 4 + 0);
        bits |= (m4.y != 0u ? 1u : 0u) << (k * 4 + 1);
        bits |= (m4.z != 0u ? 1u : 0u) << (k * 4 + 2);
        bits |= (m4.w != 0u ? 1u : 0u) << (k * 4 + 3);
      }
      Ms[r * 32 + hcol * 16 + i] = bits;
    }
  }
  __syncthreads();

  // Q A-fragments for K-dim chunks 0..31 and 32..63 (held in registers)
  FragBF qf[2];
  {
    const unsigned short* p = &Qs[(w * 16 + l16) * 72];
#pragma unroll
    for (int c = 0; c < 2; c++) {
      qf[c].u[0] = *(const uint4*)(p + c * 32 + hf * 8);
      qf[c].u[1] = *(const uint4*)(p + c * 32 + 16 + hf * 8);
    }
  }

  float mrow[8], lrow[8];
  v8f accO[4] = {};
#pragma unroll
  for (int j = 0; j < 8; j++) { mrow[j] = -3.0e38f; lrow[j] = 0.f; }

  // staging indices + K/V register prefetch of tile 0
  const int r_st  = tid >> 3;          // 0..31 (+32 per i)
  const int c8_st = (tid & 7) * 8;
  uint4 rk[4], rv[4];
#pragma unroll
  for (int i = 0; i < 4; i++) {
    int r = r_st + i * 32;
    rk[i] = *(const uint4*)(Kg + kvbase + (size_t)r * D + c8_st);
    rv[i] = *(const uint4*)(Vg + kvbase + (size_t)r * D + c8_st);
  }

  for (int kb = 0; kb < T; kb += 128) {
    __syncthreads();
    // store prefetched K (token-major) and V^T (dim-major) tiles to LDS
#pragma unroll
    for (int i = 0; i < 4; i++) {
      int r = r_st + i * 32;
      *(uint4*)&Ks[r * 72 + c8_st] = rk[i];
      const unsigned short* vs = (const unsigned short*)&rv[i];
#pragma unroll
      for (int d = 0; d < 8; d++) Vs[(c8_st + d) * 136 + r] = vs[d];
    }
    __syncthreads();
    if (kb + 128 < T) {                // issue next tile's loads now;
#pragma unroll                          // they complete under the WMMA chain
      for (int i = 0; i < 4; i++) {
        int r = r_st + i * 32;
        rk[i] = *(const uint4*)(Kg + kvbase + (size_t)(kb + 128 + r) * D + c8_st);
        rv[i] = *(const uint4*)(Vg + kvbase + (size_t)(kb + 128 + r) * D + c8_st);
      }
    }

    // S = Q @ K^T for this 128-token tile (8 col tiles x 2 K-chunks)
    v8f s[8] = {};
#pragma unroll
    for (int t = 0; t < 8; t++) {
#pragma unroll
      for (int c = 0; c < 2; c++) {
        FragBF kf;
        const unsigned short* p = &Ks[(t * 16 + l16) * 72 + c * 32 + hf * 8];
        kf.u[0] = *(const uint4*)p;
        kf.u[1] = *(const uint4*)(p + 16);
        s[t] = __builtin_amdgcn_wmma_f32_16x16x32_bf16(
            false, qf[c].v, false, kf.v, (short)0, s[t], false, false);
      }
    }

    // mask(bit test) + online softmax (rows live in 16-lane halves)
#pragma unroll
    for (int j = 0; j < 8; j++) {
      const int r = w * 16 + j + 8 * hf;               // local query row
      uint4 mr4 = *(const uint4*)&Ms[r * 32 + (kb >> 5)];  // 128 bits of row
      unsigned int words[4] = {mr4.x, mr4.y, mr4.z, mr4.w};
      float mx = -3.0e38f;
#pragma unroll
      for (int t = 0; t < 8; t++) {
        float v = s[t][j];                             // Q pre-scaled
        unsigned int bit = ((unsigned)(t & 1) << 4) | (unsigned)l16;
        if (((words[t >> 1] >> bit) & 1u) == 0u) v = -1.0e9f;
        s[t][j] = v;
        mx = fmaxf(mx, v);
      }
#pragma unroll
      for (int off = 8; off; off >>= 1) mx = fmaxf(mx, __shfl_xor(mx, off, 16));
      float mnew = fmaxf(mrow[j], mx);
      float sc = __expf(mrow[j] - mnew);
      float sum = 0.f;
#pragma unroll
      for (int t = 0; t < 8; t++) {
        float p = __expf(s[t][j] - mnew);
        s[t][j] = p;
        sum += p;
      }
#pragma unroll
      for (int off = 8; off; off >>= 1) sum += __shfl_xor(sum, off, 16);
      lrow[j] = lrow[j] * sc + sum;
#pragma unroll
      for (int nt = 0; nt < 4; nt++) accO[nt][j] *= sc;
      mrow[j] = mnew;
    }

    // write P (bf16) to this wave's private LDS rows (no barrier needed)
#pragma unroll
    for (int t = 0; t < 8; t++)
#pragma unroll
      for (int j = 0; j < 8; j++)
        Ps[(w * 16 + j + 8 * hf) * 136 + t * 16 + l16] = f2bf(s[t][j]);

    // O += P @ V  (K = 128 tokens in 4 chunks of 32; N = 64 dims = 4 tiles)
#pragma unroll
    for (int c = 0; c < 4; c++) {
      FragBF pf;
      const unsigned short* pp = &Ps[(w * 16 + l16) * 136 + c * 32];
      pf.u[0] = *(const uint4*)(pp + hf * 8);
      pf.u[1] = *(const uint4*)(pp + 16 + hf * 8);
#pragma unroll
      for (int nt = 0; nt < 4; nt++) {
        FragBF vf;
        const unsigned short* vp = &Vs[(nt * 16 + l16) * 136 + c * 32 + hf * 8];
        vf.u[0] = *(const uint4*)vp;
        vf.u[1] = *(const uint4*)(vp + 16);
        accO[nt] = __builtin_amdgcn_wmma_f32_16x16x32_bf16(
            false, pf.v, false, vf.v, (short)0, accO[nt], false, false);
      }
    }
  }

  // epilogue: normalize by l and store bf16 at head offset
#pragma unroll
  for (int j = 0; j < 8; j++) {
    float inv = 1.f / lrow[j];
    size_t row = (size_t)b * T + q0 + w * 16 + j + 8 * hf;
#pragma unroll
    for (int nt = 0; nt < 4; nt++) {
      size_t col = (size_t)h * 64 + nt * 16 + l16;
      Og[row * D + col] = f2bf(accO[nt][j] * inv);
    }
  }
}

// ---------------------------------------------------------------------------
// y = LN(x + res), torch-style: unbiased var (ddof=1), divide by (std + eps).
// One block per row (D=1024, 256 threads x 4).
// ---------------------------------------------------------------------------
__global__ __launch_bounds__(256, 1)
void add_ln_kernel(const float* __restrict__ x, const float* __restrict__ res,
                   const float* __restrict__ alpha, const float* __restrict__ beta,
                   float* __restrict__ out, unsigned short* __restrict__ out_bf,
                   int D) {
  const int row = blockIdx.x;
  const int tid = threadIdx.x;
  const float* xr = x + (size_t)row * D;
  const float* rr = res + (size_t)row * D;
  float v[4];
  float s = 0.f, q = 0.f;
#pragma unroll
  for (int i = 0; i < 4; i++) {
    int c = tid + i * 256;
    v[i] = xr[c] + rr[c];
    s += v[i];
    q += v[i] * v[i];
  }
#pragma unroll
  for (int off = 16; off; off >>= 1) {
    s += __shfl_xor(s, off, 32);
    q += __shfl_xor(q, off, 32);
  }
  __shared__ float shs[8], shq[8];
  int w = tid >> 5, lane = tid & 31;
  if (lane == 0) { shs[w] = s; shq[w] = q; }
  __syncthreads();
  float S = 0.f, Q2 = 0.f;
#pragma unroll
  for (int i = 0; i < 8; i++) { S += shs[i]; Q2 += shq[i]; }
  float mean = S / D;
  float var  = (Q2 - S * mean) / (D - 1);     // unbiased (ddof=1)
  float inv  = 1.f / (sqrtf(fmaxf(var, 0.f)) + 1e-12f);
#pragma unroll
  for (int i = 0; i < 4; i++) {
    int c = tid + i * 256;
    float y = alpha[c] * (v[i] - mean) * inv + beta[c];
    out[(size_t)row * D + c] = y;
    if (out_bf) out_bf[(size_t)row * D + c] = f2bf(y);
  }
}

// ---------------------------------------------------------------------------
// Host orchestration
// ---------------------------------------------------------------------------
extern "C" void kernel_launch(void* const* d_in, const int* in_sizes, int n_in,
                              void* d_out, int out_size, void* d_ws, size_t ws_size,
                              hipStream_t stream) {
  (void)in_sizes; (void)n_in; (void)out_size; (void)ws_size;
  constexpr int B = 2, T = 1024, D = 1024, H = 16, F = 4096;
  constexpr int BT = B * T;
  constexpr size_t AD = (size_t)BT * D;      // activation elems
  constexpr size_t WD = (size_t)D * D;       // square weight elems
  constexpr size_t FD = (size_t)F * D;

  const float* dec = (const float*)d_in[0];
  const float* enc = (const float*)d_in[1];
  const int* trg_mask = (const int*)d_in[2];
  const int* src_mask = (const int*)d_in[3];
  const float* wq1 = (const float*)d_in[4];
  const float* wk1 = (const float*)d_in[5];
  const float* wv1 = (const float*)d_in[6];
  const float* wo1 = (const float*)d_in[7];
  const float* wq2 = (const float*)d_in[8];
  const float* wk2 = (const float*)d_in[9];
  const float* wv2 = (const float*)d_in[10];
  const float* wo2 = (const float*)d_in[11];
  const float* ln1_a = (const float*)d_in[12];
  const float* ln1_b = (const float*)d_in[13];
  const float* ln2_a = (const float*)d_in[14];
  const float* ln2_b = (const float*)d_in[15];
  const float* ln3_a = (const float*)d_in[16];
  const float* ln3_b = (const float*)d_in[17];
  const float* ffn_w1 = (const float*)d_in[18];
  const float* ffn_b1 = (const float*)d_in[19];
  const float* ffn_w2 = (const float*)d_in[20];
  const float* ffn_b2 = (const float*)d_in[21];

  // bump allocator on workspace
  char* ws = (char*)d_ws;
  size_t off = 0;
  auto alloc = [&](size_t bytes) -> void* {
    void* p = ws + off;
    off = (off + bytes + 255) & ~(size_t)255;
    return p;
  };
  typedef unsigned short u16;
  u16* dec_bf = (u16*)alloc(AD * 2);
  u16* enc_bf = (u16*)alloc(AD * 2);
  u16* wbf[8];
  for (int i = 0; i < 8; i++) wbf[i] = (u16*)alloc(WD * 2);
  u16* w_ffn1_bf = (u16*)alloc(FD * 2);
  u16* w_ffn2_bf = (u16*)alloc(FD * 2);
  u16* q_bf   = (u16*)alloc(AD * 2);
  u16* k_bf   = (u16*)alloc(AD * 2);
  u16* v_bf   = (u16*)alloc(AD * 2);
  u16* attn_bf = (u16*)alloc(AD * 2);
  float* x1   = (float*)alloc(AD * 4);
  u16* x1_bf  = (u16*)alloc(AD * 2);
  float* x2   = (float*)alloc(AD * 4);
  u16* x2_bf  = (u16*)alloc(AD * 2);
  float* tmp  = (float*)alloc(AD * 4);
  u16* h_bf   = (u16*)alloc((size_t)BT * F * 2);

  auto cvt = [&](const float* src, u16* dst, size_t n) {
    int n4 = (int)(n / 4);
    cvt_f32_bf16<<<dim3((n4 + 255) / 256), dim3(256), 0, stream>>>(src, dst, n4);
  };

  // 1) convert activations + weights to bf16
  cvt(dec, dec_bf, AD);
  cvt(enc, enc_bf, AD);
  const float* wsrc[8] = {wq1, wk1, wv1, wo1, wq2, wk2, wv2, wo2};
  for (int i = 0; i < 8; i++) cvt(wsrc[i], wbf[i], WD);
  cvt(ffn_w1, w_ffn1_bf, FD);
  cvt(ffn_w2, w_ffn2_bf, FD);

  const dim3 blk(256);
  const dim3 gP(D / 128, BT / 128);       // projection GEMMs  (N=1024)
  const dim3 gF1(F / 128, BT / 128);      // FFN up            (N=4096)
  const dim3 gF2(D / 128, BT / 128);      // FFN down          (N=1024)
  const dim3 gAttn(T / 128, B * H);
  const size_t smem_attn =
      (size_t)(128 * 72 + 128 * 72 + 64 * 136 + 128 * 136) * 2 + 128 * 32 * 4;
  const float qscale = 0.125f;            // 1/sqrt(dk), folded into Q proj

  // 2) self-attention: Q/K/V projections (bf16 out)
  gemm_bt_kernel<false, false, true><<<gP, blk, 0, stream>>>(dec_bf, D, wbf[0], D, nullptr, q_bf, D, D, qscale);
  gemm_bt_kernel<false, false, true><<<gP, blk, 0, stream>>>(dec_bf, D, wbf[1], D, nullptr, k_bf, D, D, 1.f);
  gemm_bt_kernel<false, false, true><<<gP, blk, 0, stream>>>(dec_bf, D, wbf[2], D, nullptr, v_bf, D, D, 1.f);
  attn_fused_kernel<<<gAttn, blk, smem_attn, stream>>>(q_bf, k_bf, v_bf, trg_mask, attn_bf, T, D, H);
  gemm_bt_kernel<false, false, false><<<gP, blk, 0, stream>>>(attn_bf, D, wbf[3], D, nullptr, tmp, D, D, 1.f);
  add_ln_kernel<<<dim3(BT), blk, 0, stream>>>(tmp, dec, ln1_a, ln1_b, x1, x1_bf, D);

  // 3) cross-attention (query = dec, faithful to reference)
  gemm_bt_kernel<false, false, true><<<gP, blk, 0, stream>>>(dec_bf, D, wbf[4], D, nullptr, q_bf, D, D, qscale);
  gemm_bt_kernel<false, false, true><<<gP, blk, 0, stream>>>(enc_bf, D, wbf[5], D, nullptr, k_bf, D, D, 1.f);
  gemm_bt_kernel<false, false, true><<<gP, blk, 0, stream>>>(enc_bf, D, wbf[6], D, nullptr, v_bf, D, D, 1.f);
  attn_fused_kernel<<<gAttn, blk, smem_attn, stream>>>(q_bf, k_bf, v_bf, src_mask, attn_bf, T, D, H);
  gemm_bt_kernel<false, false, false><<<gP, blk, 0, stream>>>(attn_bf, D, wbf[7], D, nullptr, tmp, D, D, 1.f);
  add_ln_kernel<<<dim3(BT), blk, 0, stream>>>(tmp, x1, ln2_a, ln2_b, x2, x2_bf, D);

  // 4) FFN
  gemm_bt_kernel<true, true, true><<<gF1, blk, 0, stream>>>(x2_bf, D, w_ffn1_bf, D, ffn_b1, h_bf, F, D, 1.f);
  gemm_bt_kernel<true, false, false><<<gF2, blk, 0, stream>>>(h_bf, F, w_ffn2_bf, F, ffn_b2, tmp, D, F, 1.f);
  add_ln_kernel<<<dim3(BT), blk, 0, stream>>>(tmp, x2, ln3_a, ln3_b, (float*)d_out, nullptr, D);
}
